// MultiAxisAttention_44160853737623
// MI455X (gfx1250) — compile-verified, matrix operand order checked by reading, not run on hardware
//
#include <hip/hip_runtime.h>
#include <hip/hip_bf16.h>

// Problem constants (fixed by the reference)
constexpr int NB  = 2;      // batch
constexpr int SEQ = 2048;   // sequence length
constexpr int DM  = 1024;   // model dim
constexpr int NH  = 16;     // heads
constexpr int HD  = 64;     // head dim
constexpr int MT  = NB * SEQ;   // 4096 rows of the "token" GEMMs
constexpr float NEGV = -1e9f;

typedef __attribute__((ext_vector_type(16))) __bf16 v16bf;
typedef __attribute__((ext_vector_type(8)))  float  v8f;
typedef __attribute__((ext_vector_type(4)))  int    v4i;
typedef __attribute__((address_space(1))) v4i* gv4p;   // global int4*
typedef __attribute__((address_space(3))) v4i* lv4p;   // LDS int4*

union BFrag { v16bf v; unsigned short h[16]; unsigned int u[8]; };

// --- gfx1250 async copy global->LDS (guarded; falls back to sync copy) ------
#if defined(__has_builtin)
#if __has_builtin(__builtin_amdgcn_global_load_async_to_lds_b128)
#define HAVE_ASYNC 1
#endif
#endif
#ifndef HAVE_ASYNC
#define HAVE_ASYNC 0
#endif

#if HAVE_ASYNC
#if __has_builtin(__builtin_amdgcn_s_wait_asynccnt)
#define WAIT_ASYNC(n) __builtin_amdgcn_s_wait_asynccnt(n)
#else
#define WAIT_ASYNC(n) asm volatile("s_wait_asynccnt " #n ::: "memory")
#endif
#else
#define WAIT_ASYNC(n) ((void)0)
#endif

// float -> bf16 bits, round-to-nearest-even
__device__ __forceinline__ unsigned short f2bf(float f) {
  unsigned int u = __float_as_uint(f);
  u += 0x7FFFu + ((u >> 16) & 1u);
  return (unsigned short)(u >> 16);
}

// Load one 16-half WMMA A/B fragment (16x32 / 32x16 bf16) for this lane.
// base = this lane's row (A: row m = lane%16; B: col n = lane%16), 32
// consecutive bf16 along K. koff = 8*(lane>>4) per the ISA VGPR layout:
//   VGPR g<4  : K = 2g+{0,1} + koff ;  VGPR g>=4 : K = 16+2(g-4)+{0,1} + koff
__device__ __forceinline__ v16bf load_frag16(const unsigned short* base, int koff) {
  BFrag f;
#pragma unroll
  for (int g = 0; g < 8; ++g) {
    const int k = ((g < 4) ? (2 * g) : (16 + 2 * (g - 4))) + koff;
    f.u[g] = *(const unsigned int*)(base + k);   // k even -> dword aligned
  }
  return f.v;
}

// Copy one 128x32-half tile (global, row stride DM halves) into an LDS tile
// (row stride 56 halves, every 16B chunk 16B-aligned). 256 threads, 2 B128
// transfers each: async on CDNA5, sync fallback otherwise.
constexpr int LDST = 56;            // LDS tile row stride in halves (112B)
constexpr int TILEH = 128 * LDST;   // halves per LDS tile buffer

__device__ __forceinline__ void tile_copy(const unsigned short* __restrict__ gsrc,
                                          unsigned short* __restrict__ ldst,
                                          int tid) {
#pragma unroll
  for (int rep = 0; rep < 2; ++rep) {
    const int i   = tid + rep * 256;   // 0..511
    const int row = i >> 2;            // 0..127
    const int c   = (i & 3) * 8;       // half offset 0,8,16,24
#if HAVE_ASYNC
    __builtin_amdgcn_global_load_async_to_lds_b128(
        (gv4p)(gsrc + (size_t)row * DM + c),
        (lv4p)(ldst + row * LDST + c), 0, 0);
#else
    *(uint4*)(ldst + row * LDST + c) = *(const uint4*)(gsrc + (size_t)row * DM + c);
#endif
  }
}

// -------------------------------------------------------------------------
// Kernel 0: fp32 -> bf16 cast (x, Wqkv, Wout), L2-resident, trivial cost.
// -------------------------------------------------------------------------
__global__ __launch_bounds__(256) void cast_bf16_kernel(
    const float* __restrict__ src, unsigned short* __restrict__ dst, int n) {
  const int i = (blockIdx.x * 256 + threadIdx.x) * 4;
  if (i + 3 < n) {
    const float4 v = *(const float4*)(src + i);
    unsigned short t[4] = { f2bf(v.x), f2bf(v.y), f2bf(v.z), f2bf(v.w) };
    *(uint2*)(dst + i) = *(const uint2*)t;
  }
}

// -------------------------------------------------------------------------
// Kernel 1: QKV projection.  qkv[m,n] = sum_k x[m,k]*Wqkv[n,k] + bqkv[n]
// M=4096, N=3072, K=1024 (bf16 operands, f32 accum). Double-buffered async
// LDS staging. Scatter to Q,K [B,H,S,hd] and V^T [B,H,hd,S] (bf16).
// -------------------------------------------------------------------------
__global__ __launch_bounds__(256) void qkv_proj_kernel(
    const unsigned short* __restrict__ xb, const unsigned short* __restrict__ Wb,
    const float* __restrict__ bqkv,
    unsigned short* __restrict__ Qb, unsigned short* __restrict__ Kb,
    unsigned short* __restrict__ Vt)
{
  __shared__ unsigned short ldsA[2 * TILEH];
  __shared__ unsigned short ldsB[2 * TILEH];

  const int tid  = threadIdx.x;
  const int lane = tid & 31;
  const int wave = tid >> 5;
  const int wm   = wave >> 2;          // 0..1 : 64-row strip
  const int wn   = wave & 3;           // 0..3 : 32-col strip
  const int m0   = blockIdx.y * 128;
  const int n0   = blockIdx.x * 128;
  const int ln   = lane & 15;
  const int koff = (lane >> 4) * 8;

  const unsigned short* Ag = xb + (size_t)m0 * DM;
  const unsigned short* Bg = Wb + (size_t)n0 * DM;

  v8f acc[4][2] = {};

  tile_copy(Ag, ldsA, tid);
  tile_copy(Bg, ldsB, tid);

  constexpr int NT = DM / 32;
  for (int kt = 0; kt < NT; ++kt) {
    if (kt + 1 < NT) {
      const int nb = (kt + 1) & 1;
      tile_copy(Ag + (kt + 1) * 32, ldsA + nb * TILEH, tid);
      tile_copy(Bg + (kt + 1) * 32, ldsB + nb * TILEH, tid);
      WAIT_ASYNC(4);                 // previous tile's 4 transfers complete
    } else {
      WAIT_ASYNC(0);
    }
    __syncthreads();

    const unsigned short* a = ldsA + (kt & 1) * TILEH;
    const unsigned short* b = ldsB + (kt & 1) * TILEH;
    v16bf aF[4];
#pragma unroll
    for (int mt = 0; mt < 4; ++mt)
      aF[mt] = load_frag16(a + (wm * 64 + mt * 16 + ln) * LDST, koff);
#pragma unroll
    for (int nt = 0; nt < 2; ++nt) {
      const v16bf bF = load_frag16(b + (wn * 32 + nt * 16 + ln) * LDST, koff);
#pragma unroll
      for (int mt = 0; mt < 4; ++mt)
        acc[mt][nt] = __builtin_amdgcn_wmma_f32_16x16x32_bf16(
            false, aF[mt], false, bF, (short)0, acc[mt][nt], false, false);
    }
    __syncthreads();                 // all reads done before buffer reuse
  }

  const int halfsel = lane >> 4;
#pragma unroll
  for (int mt = 0; mt < 4; ++mt) {
#pragma unroll
    for (int nt = 0; nt < 2; ++nt) {
      const int n = n0 + wn * 32 + nt * 16 + ln;
      const float bias = bqkv[n];
      const int which = n >> 10;               // 0=Q 1=K 2=V
      const int hh    = (n & 1023) >> 6;
      const int dd    = n & 63;
#pragma unroll
      for (int r = 0; r < 8; ++r) {
        const int m = m0 + wm * 64 + mt * 16 + r + 8 * halfsel;
        const int b = m >> 11;                 // / SEQ
        const int s = m & (SEQ - 1);
        const unsigned short val = f2bf(acc[mt][nt][r] + bias);
        if (which == 0)
          Qb[(((size_t)(b * NH + hh)) * SEQ + s) * HD + dd] = val;
        else if (which == 1)
          Kb[(((size_t)(b * NH + hh)) * SEQ + s) * HD + dd] = val;
        else
          Vt[(((size_t)(b * NH + hh)) * HD + dd) * SEQ + s] = val;
      }
    }
  }
}

// -------------------------------------------------------------------------
// Kernel 2: flash attention. One wave per 16-row q-tile, key blocks of 32.
// Online softmax in f32; P re-staged through per-wave LDS (D-layout -> A-layout).
// -------------------------------------------------------------------------
__global__ __launch_bounds__(256) void attn_kernel(
    const unsigned short* __restrict__ Qb, const unsigned short* __restrict__ Kb,
    const unsigned short* __restrict__ Vt, const int* __restrict__ seq_lens,
    unsigned short* __restrict__ attnb)
{
  __shared__ unsigned short ldsP[8][16 * 36];   // per-wave 16x32 P tile (+pad)

  const int lane    = threadIdx.x & 31;
  const int wave    = threadIdx.x >> 5;
  const int b       = blockIdx.z;
  const int h       = blockIdx.y;
  const int q0      = blockIdx.x * 128 + wave * 16;
  const int seqlen  = seq_lens[b];
  const int halfsel = lane >> 4;
  const int koff    = halfsel * 8;
  const int ln      = lane & 15;
  const float scale = 0.125f;                   // 1/sqrt(64)

  // Q fragments (16 x 64 = 2 K-steps), loaded once
  const unsigned short* Qrow =
      Qb + (((size_t)(b * NH + h)) * SEQ + (q0 + ln)) * HD;
  v16bf aQ[2];
  aQ[0] = load_frag16(Qrow, koff);
  aQ[1] = load_frag16(Qrow + 32, koff);

  v8f oc[4] = {};                               // O accumulators 16x64
  float mrow[8], lrow[8];
#pragma unroll
  for (int r = 0; r < 8; ++r) { mrow[r] = -__builtin_inff(); lrow[r] = 0.f; }

  unsigned short* Pw = &ldsP[wave][0];
  const int nkb = (q0 + 16 + 31) >> 5;          // causal bound on key blocks

  for (int kb = 0; kb < nkb; ++kb) {
    const int kbase = kb * 32;

    // S = Q K^T for a 16x32 block of scores (2 n-tiles x 2 d-steps)
    v8f sc[2] = {};
#pragma unroll
    for (int nt = 0; nt < 2; ++nt) {
      const unsigned short* Krow =
          Kb + (((size_t)(b * NH + h)) * SEQ + (kbase + nt * 16 + ln)) * HD;
      const v16bf bK0 = load_frag16(Krow, koff);
      const v16bf bK1 = load_frag16(Krow + 32, koff);
      sc[nt] = __builtin_amdgcn_wmma_f32_16x16x32_bf16(
          false, aQ[0], false, bK0, (short)0, sc[nt], false, false);
      sc[nt] = __builtin_amdgcn_wmma_f32_16x16x32_bf16(
          false, aQ[1], false, bK1, (short)0, sc[nt], false, false);
    }

    // scale + additive mask + online softmax (rows live in 16-lane groups)
    float p[2][8];
#pragma unroll
    for (int r = 0; r < 8; ++r) {
      const int qpos = q0 + r + 8 * halfsel;
      float vmax = -__builtin_inff();
#pragma unroll
      for (int nt = 0; nt < 2; ++nt) {
        const int kpos = kbase + nt * 16 + ln;
        float sval = sc[nt][r] * scale;
        if (kpos > qpos)    sval += NEGV;       // causal
        if (kpos >= seqlen) sval += NEGV;       // padding
        p[nt][r] = sval;
        vmax = fmaxf(vmax, sval);
      }
#pragma unroll
      for (int msk = 1; msk < 16; msk <<= 1)
        vmax = fmaxf(vmax, __shfl_xor(vmax, msk));
      const float newmax = fmaxf(mrow[r], vmax);
      const float alpha  = expf(mrow[r] - newmax);
      float rsum = 0.f;
#pragma unroll
      for (int nt = 0; nt < 2; ++nt) {
        const float e = expf(p[nt][r] - newmax);
        p[nt][r] = e;
        rsum += e;
      }
#pragma unroll
      for (int msk = 1; msk < 16; msk <<= 1)
        rsum += __shfl_xor(rsum, msk);
      lrow[r] = lrow[r] * alpha + rsum;
      mrow[r] = newmax;
#pragma unroll
      for (int j = 0; j < 4; ++j) oc[j][r] *= alpha;
    }

    // D-layout P -> LDS -> A-layout fragment (same-wave DS ops are in order)
#pragma unroll
    for (int r = 0; r < 8; ++r) {
      const int m = r + 8 * halfsel;
      Pw[m * 36 + ln]      = f2bf(p[0][r]);
      Pw[m * 36 + 16 + ln] = f2bf(p[1][r]);
    }
    asm volatile("s_wait_dscnt 0" ::: "memory");
    const v16bf aP = load_frag16(Pw + ln * 36, koff);

    // O += P @ V  (V^T rows are contiguous along keys)
#pragma unroll
    for (int j = 0; j < 4; ++j) {
      const unsigned short* Vrow =
          Vt + (((size_t)(b * NH + h)) * HD + j * 16 + ln) * SEQ + kbase;
      const v16bf bV = load_frag16(Vrow, koff);
      oc[j] = __builtin_amdgcn_wmma_f32_16x16x32_bf16(
          false, aP, false, bV, (short)0, oc[j], false, false);
    }
  }

  // finalize: O /= l, store bf16 into attnb [B*S, D]
#pragma unroll
  for (int r = 0; r < 8; ++r) {
    const float inv = (lrow[r] > 0.f) ? (1.f / lrow[r]) : 0.f;
#pragma unroll
    for (int j = 0; j < 4; ++j) oc[j][r] *= inv;
  }
#pragma unroll
  for (int j = 0; j < 4; ++j) {
#pragma unroll
    for (int r = 0; r < 8; ++r) {
      const int m = r + 8 * halfsel;
      attnb[((size_t)(b * SEQ) + q0 + m) * DM + h * HD + j * 16 + ln] =
          f2bf(oc[j][r]);
    }
  }
}

// -------------------------------------------------------------------------
// Kernel 3: output projection. out[m,n] = sum_k attn[m,k]*Wout[n,k] + bout[n]
// M=4096, N=1024, K=1024. Same async double-buffered staging; fp32 output.
// -------------------------------------------------------------------------
__global__ __launch_bounds__(256) void out_proj_kernel(
    const unsigned short* __restrict__ attnb, const unsigned short* __restrict__ Wb,
    const float* __restrict__ bout, float* __restrict__ out)
{
  __shared__ unsigned short ldsA[2 * TILEH];
  __shared__ unsigned short ldsB[2 * TILEH];

  const int tid  = threadIdx.x;
  const int lane = tid & 31;
  const int wave = tid >> 5;
  const int wm   = wave >> 2;
  const int wn   = wave & 3;
  const int m0   = blockIdx.y * 128;
  const int n0   = blockIdx.x * 128;
  const int ln   = lane & 15;
  const int koff = (lane >> 4) * 8;

  const unsigned short* Ag = attnb + (size_t)m0 * DM;
  const unsigned short* Bg = Wb + (size_t)n0 * DM;

  v8f acc[4][2] = {};

  tile_copy(Ag, ldsA, tid);
  tile_copy(Bg, ldsB, tid);

  constexpr int NT = DM / 32;
  for (int kt = 0; kt < NT; ++kt) {
    if (kt + 1 < NT) {
      const int nb = (kt + 1) & 1;
      tile_copy(Ag + (kt + 1) * 32, ldsA + nb * TILEH, tid);
      tile_copy(Bg + (kt + 1) * 32, ldsB + nb * TILEH, tid);
      WAIT_ASYNC(4);
    } else {
      WAIT_ASYNC(0);
    }
    __syncthreads();

    const unsigned short* a = ldsA + (kt & 1) * TILEH;
    const unsigned short* b = ldsB + (kt & 1) * TILEH;
    v16bf aF[4];
#pragma unroll
    for (int mt = 0; mt < 4; ++mt)
      aF[mt] = load_frag16(a + (wm * 64 + mt * 16 + ln) * LDST, koff);
#pragma unroll
    for (int nt = 0; nt < 2; ++nt) {
      const v16bf bF = load_frag16(b + (wn * 32 + nt * 16 + ln) * LDST, koff);
#pragma unroll
      for (int mt = 0; mt < 4; ++mt)
        acc[mt][nt] = __builtin_amdgcn_wmma_f32_16x16x32_bf16(
            false, aF[mt], false, bF, (short)0, acc[mt][nt], false, false);
    }
    __syncthreads();
  }

  const int halfsel = lane >> 4;
#pragma unroll
  for (int mt = 0; mt < 4; ++mt) {
#pragma unroll
    for (int nt = 0; nt < 2; ++nt) {
      const int n = n0 + wn * 32 + nt * 16 + ln;
      const float bias = bout[n];
#pragma unroll
      for (int r = 0; r < 8; ++r) {
        const int m = m0 + wm * 64 + mt * 16 + r + 8 * halfsel;
        out[(size_t)m * DM + n] = acc[mt][nt][r] + bias;
      }
    }
  }
}

// -------------------------------------------------------------------------
extern "C" void kernel_launch(void* const* d_in, const int* in_sizes, int n_in,
                              void* d_out, int out_size, void* d_ws, size_t ws_size,
                              hipStream_t stream) {
  (void)in_sizes; (void)n_in; (void)out_size; (void)ws_size;
  const float* x        = (const float*)d_in[0];
  const int*   seq_lens = (const int*)  d_in[1];
  const float* Wqkv     = (const float*)d_in[2];
  const float* bqkv     = (const float*)d_in[3];
  const float* Wout     = (const float*)d_in[4];
  const float* bout     = (const float*)d_in[5];
  float* out = (float*)d_out;

  const size_t per = (size_t)NB * NH * SEQ * HD;   // 4M bf16 elems
  unsigned short* Qb    = (unsigned short*)d_ws;
  unsigned short* Kb    = Qb + per;
  unsigned short* Vt    = Kb + per;
  unsigned short* attnb = Vt + per;
  unsigned short* xb    = attnb + per;                       // MT*DM
  unsigned short* Wqkvb = xb + (size_t)MT * DM;              // 3*DM*DM
  unsigned short* Woutb = Wqkvb + (size_t)3 * DM * DM;       // DM*DM

  const int nx = MT * DM, nq = 3 * DM * DM, no = DM * DM;
  cast_bf16_kernel<<<nx / 1024, 256, 0, stream>>>(x, xb, nx);
  cast_bf16_kernel<<<nq / 1024, 256, 0, stream>>>(Wqkv, Wqkvb, nq);
  cast_bf16_kernel<<<no / 1024, 256, 0, stream>>>(Wout, Woutb, no);

  qkv_proj_kernel<<<dim3(3 * DM / 128, MT / 128), 256, 0, stream>>>(
      xb, Wqkvb, bqkv, Qb, Kb, Vt);
  attn_kernel<<<dim3(SEQ / 128, NH, NB), 256, 0, stream>>>(
      Qb, Kb, Vt, seq_lens, attnb);
  out_proj_kernel<<<dim3(DM / 128, MT / 128), 256, 0, stream>>>(
      attnb, Woutb, bout, out);
}
